// SANLayer_72894184948136
// MI455X (gfx1250) — compile-verified, matrix operand order checked by reading, not run on hardware
//
#include <hip/hip_runtime.h>
#include <math.h>

// ---------------------------------------------------------------------------
// SAN layer for MI455X (gfx1250, wave32, WMMA).
// N=8192 cells, DEG=32 edges/row, CIN=COUT=128, NF=2 filters.
//
// Dominant cost: z = P @ (x @ W_har), P is 8192x8192 fp32 (256 MB) -> HBM
// roofline ~11us @ 23.3 TB/s. bf16 WMMA (fp32 accumulate) keeps the GEMM
// memory-bound. Split-K=4 into separate partial buffers (deterministic),
// reduced in the final aggregation kernel.
// Round-4 tuning: __builtin_amdgcn_sched_barrier(0) fences pin the
// load(j+2) | wmma(j) interleave so the scheduler cannot collapse the
// 3-deep B-fragment rotation into one register octet (rounds 2/3 regressed
// to ds_load; s_wait_dscnt 0x0; v_wmma per step).
// ---------------------------------------------------------------------------

typedef __attribute__((ext_vector_type(16))) __bf16 v16bf;
typedef __attribute__((ext_vector_type(8)))  __bf16 v8bf;
typedef __attribute__((ext_vector_type(4)))  __bf16 v4bf;
typedef __attribute__((ext_vector_type(8)))  float  v8f;

#define SAN_N    8192
#define SAN_C    128
#define SAN_DEG  32
#define SAN_E    (SAN_N * SAN_DEG)

#define TILE_M   128
#define KCHUNK   64
#define LDSP     72      // padded K-stride (bf16 units); 144B rows, 16B aligned

// Load a 16-element bf16 WMMA fragment from LDS.
// Layout: element e of lane (r,h) sits at K = (e>>3)*16 + h*8 + (e&7),
// i.e. two contiguous 8-bf16 (16B) groups, 32B apart.
__device__ __forceinline__ v16bf san_load_frag(const __bf16* p) {
    const v8bf lo = *(const v8bf*)(p);
    const v8bf hi = *(const v8bf*)(p + 16);
    v16bf f;
#pragma unroll
    for (int e = 0; e < 8; ++e) { f[e] = lo[e]; f[e + 8] = hi[e]; }
    return f;
}

// ---------------------------------------------------------------------------
// Generic bf16 WMMA GEMM: C[part] = A[:, kslice] * B[kslice, 128]
//   grid.x : M/128 tiles, grid.y : K splits
//   A fp32 row-major (lda), B fp32 row-major [K x 128], C fp32 [8192 x 128]
// 8 waves; wave w owns output rows 16w..16w+15, all 8 column subtiles.
// ---------------------------------------------------------------------------
__global__ __launch_bounds__(256)
void san_wmma_gemm(const float* __restrict__ A, const float* __restrict__ B,
                   float* __restrict__ C, int lda, int kPerSplit, int partStride)
{
    __shared__ __bf16 ldsA[TILE_M * LDSP];   // [row][k]  (bf16)
    __shared__ __bf16 ldsB[SAN_C  * LDSP];   // [col][k]  (bf16, B transposed)

    const int t    = threadIdx.x;
    const int wave = t >> 5;
    const int lane = t & 31;
    const int r    = lane & 15;       // row-in-subtile / B column
    const int h    = lane >> 4;       // half-wave selector
    const int mBase = blockIdx.x * TILE_M;
    const int kBase = blockIdx.y * kPerSplit;
    float* Cpart = C + (size_t)blockIdx.y * (size_t)partStride;

    v8f acc[8];
#pragma unroll
    for (int j = 0; j < 8; ++j)
#pragma unroll
        for (int e = 0; e < 8; ++e) acc[j][e] = 0.0f;

    const int nK = kPerSplit / KCHUNK;
    for (int kt = 0; kt < nK; ++kt) {
        const int k0 = kBase + kt * KCHUNK;
        __syncthreads();   // protect LDS from previous iteration's readers

        // --- Stage A tile: 128 rows x 64 k, fp32 -> packed bf16 (b64 stores)
#pragma unroll
        for (int i = 0; i < 8; ++i) {
            const int idx = t + i * 256;        // float4 units
            const int row = idx >> 4;           // 0..127
            const int c4  = idx & 15;           // 0..15
            const float4 v = *(const float4*)(A + (size_t)(mBase + row) * lda + k0 + c4 * 4);
            const v4bf pk = { (__bf16)v.x, (__bf16)v.y, (__bf16)v.z, (__bf16)v.w };
            *(v4bf*)&ldsA[row * LDSP + c4 * 4] = pk;
        }

        // --- Stage B tile transposed: read 4 consecutive K-rows of the same
        //     4 columns so each column's 4 bf16 K-values pack into one b64.
#pragma unroll
        for (int o = 0; o < 2; ++o) {
            const int idx = t + o * 256;        // 0..511
            const int n4  = idx & 31;           // column group (4 cols)
            const int kb  = (idx >> 5) * 4;     // 0,4,...,60
            float4 rr[4];
#pragma unroll
            for (int k = 0; k < 4; ++k)
                rr[k] = *(const float4*)(B + (size_t)(k0 + kb + k) * SAN_C + n4 * 4);
            const float* pr = (const float*)rr; // pr[k*4 + q]
#pragma unroll
            for (int q = 0; q < 4; ++q) {
                const v4bf pk = { (__bf16)pr[0 * 4 + q], (__bf16)pr[1 * 4 + q],
                                  (__bf16)pr[2 * 4 + q], (__bf16)pr[3 * 4 + q] };
                *(v4bf*)&ldsB[(n4 * 4 + q) * LDSP + kb] = pk;
            }
        }

        // --- Prefetch next K-chunk (1 cacheline per thread per operand)
        if (kt + 1 < nK) {
            const int k1 = k0 + KCHUNK;
            __builtin_prefetch(A + (size_t)(mBase + (t >> 1)) * lda + k1 + (t & 1) * 32, 0, 1);
            __builtin_prefetch(B + (size_t)(k1 + (t >> 2)) * SAN_C + (t & 3) * 32, 0, 1);
        }
        __syncthreads();

        // --- K-inner: two 16x16x32 WMMA steps. B fragments flow through a
        //     3-deep register rotation, loads 2 fragments ahead of use.
        //     sched_barrier(0) fences forbid the scheduler from sinking the
        //     loads back to their consumers, forcing 3 live octets and
        //     s_wait_dscnt 4 (two fragment fetches in flight per WMMA).
#pragma unroll
        for (int kc = 0; kc < KCHUNK; kc += 32) {
            const v16bf afr = san_load_frag(&ldsA[(wave * 16 + r) * LDSP + kc + h * 8]);
            const __bf16* pbBase = &ldsB[r * LDSP + kc + h * 8];
            v16bf bfr[3];
            bfr[0] = san_load_frag(pbBase);
            bfr[1] = san_load_frag(pbBase + 16 * LDSP);
#pragma unroll
            for (int j = 0; j < 8; ++j) {
                if (j + 2 < 8)
                    bfr[(j + 2) % 3] = san_load_frag(pbBase + (j + 2) * 16 * LDSP);
                __builtin_amdgcn_sched_barrier(0);   // loads stay above this fence
                acc[j] = __builtin_amdgcn_wmma_f32_16x16x32_bf16(
                    false, afr, false, bfr[j % 3], (short)0, acc[j], false, false);
                __builtin_amdgcn_sched_barrier(0);   // wmma stays before next loads
            }
        }
    }

    // Epilogue: C layout -> lane (r,h), VGPR v holds (M = v + 8h, Ncol = r)
#pragma unroll
    for (int j = 0; j < 8; ++j)
#pragma unroll
        for (int v = 0; v < 8; ++v) {
            const int row = mBase + wave * 16 + v + 8 * h;
            const int col = j * 16 + r;
            Cpart[(size_t)row * SAN_C + col] = acc[j][v];
        }
}

// ---------------------------------------------------------------------------
// Attention scores: for conv c, node i:
//   S1[c][i] = xm_c[i] . a_c[0:128]   (source term)
//   S2[c][i] = xm_c[i] . a_c[128:256] (target term)
// One wave per (c,i); lane-strided loads + xor-butterfly reduction (wave32).
// ---------------------------------------------------------------------------
__global__ __launch_bounds__(256)
void san_attn_scores(const float* __restrict__ xms,
                     const float* __restrict__ aD, const float* __restrict__ aU,
                     float* __restrict__ S1, float* __restrict__ S2)
{
    const int wave = threadIdx.x >> 5;
    const int lane = threadIdx.x & 31;
    const int gw   = blockIdx.x * 8 + wave;       // 0 .. 4*N-1
    const int c    = gw >> 13;                    // conv 0..3
    const int i    = gw & (SAN_N - 1);

    const float* xm = xms + (size_t)c * SAN_N * SAN_C + (size_t)i * SAN_C;
    const float* a  = (c & 1) ? aU : aD;          // c0,c2 -> down ; c1,c3 -> up
    a += (c >> 1) * 2 * SAN_C;                    // filter p = c>>1
    const float* a2 = a + SAN_C;

    float s1 = 0.0f, s2 = 0.0f;
#pragma unroll
    for (int q = 0; q < 4; ++q) {
        const float xv = xm[lane + 32 * q];
        s1 += xv * a[lane + 32 * q];
        s2 += xv * a2[lane + 32 * q];
    }
#pragma unroll
    for (int off = 16; off > 0; off >>= 1) {
        s1 += __shfl_xor(s1, off);
        s2 += __shfl_xor(s2, off);
    }
    if (lane == 0) {
        S1[c * SAN_N + i] = s1;
        S2[c * SAN_N + i] = s2;
    }
}

// ---------------------------------------------------------------------------
// Per-row softmax + neighbor aggregation + split-K partial reduction.
// One wave per target row i. Lane k owns edge k (DEG == 32 == wave width):
//   e_k = elu(S1[src_k] + S2[i]);  att = softmax_k(e)
//   z[i] = sum_s parts[s][i] + sum_c sum_k att_k * xm_c[src_k]
// Lane l also owns output columns 4l..4l+3 for the accumulation phase.
// ---------------------------------------------------------------------------
__global__ __launch_bounds__(256)
void san_aggregate(const float* __restrict__ parts, int partStride,
                   const float* __restrict__ xms,
                   const float* __restrict__ S1, const float* __restrict__ S2,
                   const int* __restrict__ idxD, const int* __restrict__ idxU,
                   float* __restrict__ z)
{
    const int wave = threadIdx.x >> 5;
    const int lane = threadIdx.x & 31;
    const int i    = blockIdx.x * 8 + wave;

    // seed with split-K partials of the harmonic conv (P @ xh)
    float4 acc = make_float4(0.f, 0.f, 0.f, 0.f);
#pragma unroll
    for (int s = 0; s < 4; ++s) {
        const float4 p = *(const float4*)(parts + (size_t)s * partStride
                                          + (size_t)i * SAN_C + lane * 4);
        acc.x += p.x; acc.y += p.y; acc.z += p.z; acc.w += p.w;
    }

#pragma unroll
    for (int c = 0; c < 4; ++c) {
        const int* idx = (c & 1) ? idxU : idxD;   // src row of COO: idx[E + e]
        const int  src = idx[SAN_E + i * SAN_DEG + lane];

        float e = S1[c * SAN_N + src] + S2[c * SAN_N + i];
        e = (e > 0.0f) ? e : expm1f(e);           // jax.nn.elu

        float m = e;
#pragma unroll
        for (int off = 16; off > 0; off >>= 1) m = fmaxf(m, __shfl_xor(m, off));
        const float ee = __expf(e - m);
        float d = ee;
#pragma unroll
        for (int off = 16; off > 0; off >>= 1) d += __shfl_xor(d, off);
        const float att = ee / d;

        const float* xm = xms + (size_t)c * SAN_N * SAN_C;
#pragma unroll 8
        for (int k = 0; k < SAN_DEG; ++k) {
            const float a  = __shfl(att, k);
            const int   sk = __shfl(src, k);
            const float4 v = *(const float4*)(xm + (size_t)sk * SAN_C + lane * 4);
            acc.x += a * v.x; acc.y += a * v.y; acc.z += a * v.z; acc.w += a * v.w;
        }
    }
    *(float4*)(z + (size_t)i * SAN_C + lane * 4) = acc;
}

// ---------------------------------------------------------------------------
extern "C" void kernel_launch(void* const* d_in, const int* in_sizes, int n_in,
                              void* d_out, int out_size, void* d_ws, size_t ws_size,
                              hipStream_t stream) {
    (void)in_sizes; (void)n_in; (void)out_size; (void)ws_size;

    const float* x   = (const float*)d_in[0];   // [N, CIN]
    const float* P   = (const float*)d_in[1];   // [N, N]
    const float* Wd  = (const float*)d_in[2];   // [NF, CIN, COUT]
    const float* aD  = (const float*)d_in[3];   // [NF, 2*COUT, 1]
    const float* Wu  = (const float*)d_in[4];   // [NF, CIN, COUT]
    const float* aU  = (const float*)d_in[5];   // [NF, 2*COUT, 1]
    const float* Wh  = (const float*)d_in[6];   // [CIN, COUT]
    const int*  idxD = (const int*)d_in[7];     // [2, E]
    const int*  idxU = (const int*)d_in[8];     // [2, E]
    float* z = (float*)d_out;                   // [N, COUT]

    float* ws    = (float*)d_ws;
    const int NC = SAN_N * SAN_C;               // 1<<20 elements
    float* xh    = ws;                          // harmonic projection  (1 NC)
    float* xms   = ws + (size_t)NC;             // 4 conv projections   (4 NC)
    float* parts = ws + (size_t)5 * NC;         // split-K partials     (4 NC)
    float* S1    = ws + (size_t)9 * NC;         // [4][N] src scores
    float* S2    = S1 + 4 * SAN_N;              // [4][N] tgt scores

    const dim3 blk(256);
    const dim3 gProj(SAN_N / TILE_M, 1);        // 64 WGs, K=128

    // 1) Projections: xh = x@W_har ; xm_c = x@W_c   (conv order: d0,u0,d1,u1)
    san_wmma_gemm<<<gProj, blk, 0, stream>>>(x, Wh,               xh,           SAN_C, SAN_C, 0);
    san_wmma_gemm<<<gProj, blk, 0, stream>>>(x, Wd,               xms + 0 * NC, SAN_C, SAN_C, 0);
    san_wmma_gemm<<<gProj, blk, 0, stream>>>(x, Wu,               xms + 1 * NC, SAN_C, SAN_C, 0);
    san_wmma_gemm<<<gProj, blk, 0, stream>>>(x, Wd + SAN_C*SAN_C, xms + 2 * NC, SAN_C, SAN_C, 0);
    san_wmma_gemm<<<gProj, blk, 0, stream>>>(x, Wu + SAN_C*SAN_C, xms + 3 * NC, SAN_C, SAN_C, 0);

    // 2) Per-node attention scores (one wave per (conv, node))
    san_attn_scores<<<dim3(4 * SAN_N / 8), blk, 0, stream>>>(xms, aD, aU, S1, S2);

    // 3) Harmonic conv: parts[s] = P[:, sK:(s+1)K] @ xh[sK:(s+1)K, :], split-K=4
    san_wmma_gemm<<<dim3(SAN_N / TILE_M, 4), blk, 0, stream>>>(P, xh, parts,
                                                               SAN_N, SAN_N / 4, NC);

    // 4) Softmax + gather-aggregate + partial reduction -> z
    san_aggregate<<<dim3(SAN_N / 8), blk, 0, stream>>>(parts, NC, xms, S1, S2,
                                                       idxD, idxU, z);
}
